// NeuralImplicitComputationGraph_19593640804632
// MI455X (gfx1250) — matrix operand order repaired and checked by reading, not compiled, not run on hardware
//
#include <hip/hip_runtime.h>
#include <math.h>

// ---------------------------------------------------------------------------
// NeuralImplicitComputationGraph sequential scan for MI455X (gfx1250).
// Persistent-kernel design: 32 workgroups x 256 threads (8 wave32s each),
// device-wide sense barrier per pipeline stage. All matvecs use
// v_wmma_f32_16x16x32_bf16 with row-replicated A tiles; weights are
// pre-swizzled once into WMMA B-fragment order (bf16) so each lane's B
// fragment is one contiguous 32-byte load (L2-resident: ~19.7MB << 192MB L2).
// ---------------------------------------------------------------------------

typedef __attribute__((ext_vector_type(16))) __bf16 v16bf;
typedef __attribute__((ext_vector_type(8)))  float  v8f;

#define L_STEPS   8192
#define H_DIM     1024
#define E_DIM     4224
#define NWG       32
#define NTHR      256
#define TOTWAVES  (NWG * (NTHR / 32))
#define KT        32            // K chunks of 32 (K = 1024 for every matvec)

// d_ws layout (bytes)
//   [0, 256)        control: 0=bar_count 1=bar_gen 2=last_e 3=last_f
//                            4-5=e_key(u64) 6-7=f_key(u64) 8=sumsq(f32)
//   [256, +4096)    h buffer 0 (1024 f32)
//   [+4096, +4096)  h buffer 1
//   [+, 1024)       rec buffer (256 f32)
//   [+, 1024)       rec_new buffer (256 f32)
//   [10496, ...)    bf16 weights in WMMA-fragment order (~19.7 MB)
#define WT_BYTE_OFF   10496
#define OFF_WIN       0
#define OFF_WS        1048576
#define OFF_WEDGE     5242880
#define OFF_WFN       9568256
#define OFF_WREC      9584640

// ------------------------- device helpers ----------------------------------

__device__ __forceinline__ float gelu_exact(float v) {
    return 0.5f * v * (1.0f + erff(v * 0.70710678118654752f));
}

__device__ __forceinline__ float rng_uniform(unsigned long long s) {
    s += 0x9E3779B97F4A7C15ull;
    s = (s ^ (s >> 30)) * 0xBF58476D1CE4E5B9ull;
    s = (s ^ (s >> 27)) * 0x94D049BB133111EBull;
    s ^= s >> 31;
    return ((float)(s >> 40) + 0.5f) * (1.0f / 16777216.0f);
}

__device__ __forceinline__ float gumbel_noise(int step, int salt, int col) {
    unsigned long long s = ((unsigned long long)(unsigned)salt << 40) ^
                           ((unsigned long long)(unsigned)step << 16) ^
                           (unsigned long long)(unsigned)col;
    float u = rng_uniform(s);
    return -logf(-logf(u));
}

// pack (value, col) so atomicMax picks max value, ties -> smallest col
__device__ __forceinline__ unsigned long long pack_key(float v, int col) {
    unsigned b = __float_as_uint(v);
    unsigned ob = (b & 0x80000000u) ? ~b : (b | 0x80000000u);
    return ((unsigned long long)ob << 32) |
           (unsigned long long)(0xFFFFFFFFu - (unsigned)col);
}

// Device-wide sense barrier (all NWG blocks co-resident).
__device__ __forceinline__ void global_barrier(unsigned* ctrl) {
    __syncthreads();
    __threadfence();                      // release prior global writes
    if (threadIdx.x == 0) {
        volatile unsigned* gen = ctrl + 1;
        unsigned g = *gen;
        if (atomicAdd(ctrl, 1u) == (unsigned)(gridDim.x - 1)) {
            atomicExch(ctrl, 0u);         // reset arrival count
            __threadfence();
            atomicAdd(ctrl + 1, 1u);      // flip generation
        } else {
            while (*gen == g) { __builtin_amdgcn_s_sleep(1); }
        }
    }
    __syncthreads();
    __threadfence();                      // acquire: see other WGs' writes
}

// Build the row-replicated A operand in WMMA A-fragment order in LDS.
// A-frag elem e of lane ln maps to K = e + 16*(ln>>4).
template <typename F>
__device__ __forceinline__ void build_A(__bf16* A_lds, F src) {
    for (int d = threadIdx.x; d < H_DIM; d += NTHR) {
        int e  = d & 15;
        int ln = (d >> 4) & 31;
        int kt = d >> 9;
        int k  = kt * 32 + e + ((ln >> 4) << 4);
        A_lds[d] = (__bf16)src(k);
    }
}

// One 16-column output tile of a K=1024 GEMV: 32 chained bf16 WMMAs.
// Result for column (nt*16 + lane) lives in acc[0] of lanes 0..15.
__device__ __forceinline__ v8f matvec_tile(const __bf16* Wt, int nt, int lane,
                                           const __bf16* A_lds) {
    v8f acc = {};
#pragma unroll 8
    for (int kt = 0; kt < KT; ++kt) {
        v16bf a = *(const v16bf*)(A_lds + (kt << 9) + (lane << 4));
        v16bf b = *(const v16bf*)(Wt + (((size_t)(nt * KT + kt) * 32) + lane) * 16);
        acc = __builtin_amdgcn_wmma_f32_16x16x32_bf16(
            false, a, false, b, (short)0, acc, false, false);
    }
    return acc;
}

// ------------------------- setup kernels -----------------------------------

// Convert one fp32 K=1024 x N matrix into bf16 WMMA B-fragment order:
// dst[((nt*KT + kt)*32 + lane)*16 + e] = src[(kt*32 + e + 16*(lane>>4))*N
//                                            + nt*16 + (lane&15)]
__global__ void convert_tiled(const float* __restrict__ src,
                              __bf16* __restrict__ dst, int N, int total) {
    int d = blockIdx.x * blockDim.x + threadIdx.x;
    if (d >= total) return;
    int e  = d & 15;
    int ln = (d >> 4) & 31;
    int t  = d >> 9;
    int kt = t & (KT - 1);
    int nt = t >> 5;
    int k  = kt * 32 + e + ((ln >> 4) << 4);
    int n  = nt * 16 + (ln & 15);
    dst[d] = (__bf16)src[(size_t)k * N + n];
}

__global__ void init_kernel(unsigned* ctrl, float* rec_buf,
                            const float* __restrict__ init_rec) {
    int t = threadIdx.x;
    if (t < 16) ctrl[t] = 0u;      // barrier, last_e/f, keys, sumsq
    rec_buf[t] = init_rec[t];      // 256 threads
}

// ------------------------- main persistent kernel --------------------------

__global__ void __launch_bounds__(NTHR)
nicg_main(const float* __restrict__ pe,     const float* __restrict__ b_in,
          const float* __restrict__ bs,     const float* __restrict__ b_edge,
          const float* __restrict__ b_fn,   const float* __restrict__ b_rec,
          const __bf16* __restrict__ Wt,    unsigned* ctrl,
          float* hb0, float* hb1, float* rec_buf, float* rec_new, float* out) {
    __shared__ __attribute__((aligned(32))) __bf16 A_lds[H_DIM];
    __shared__ float s_scale;

    const int lane = threadIdx.x & 31;
    const int gwid = blockIdx.x * (NTHR / 32) + (threadIdx.x >> 5);

    const __bf16* W_in_t   = Wt + OFF_WIN;
    const __bf16* Ws_t     = Wt + OFF_WS;
    const __bf16* W_edge_t = Wt + OFF_WEDGE;
    const __bf16* W_fn_t   = Wt + OFF_WFN;
    const __bf16* W_rec_t  = Wt + OFF_WREC;

    for (int step = 0; step < L_STEPS; ++step) {
        // ---- stage A: x = [pe[step], pe[last_e], pe[last_f], rec] ----------
        {
            int le = (int)((volatile unsigned*)ctrl)[2];
            int lf = (int)((volatile unsigned*)ctrl)[3];
            build_A(A_lds, [&](int k) -> float {
                if (k < 256)      return pe[(size_t)step * 256 + k];
                else if (k < 512) return pe[(size_t)le * 256 + (k - 256)];
                else if (k < 768) return pe[(size_t)lf * 256 + (k - 512)];
                else              return rec_buf[k - 768];
            });
            __syncthreads();
        }
        // ---- stage B: h0 = x @ W_in + b_in  (64 column tiles) --------------
        for (int nt = gwid; nt < 64; nt += TOTWAVES) {
            v8f acc = matvec_tile(W_in_t, nt, lane, A_lds);
            if (lane < 16) hb0[nt * 16 + lane] = acc[0] + b_in[nt * 16 + lane];
        }
        global_barrier(ctrl);

        // ---- stage C: 4x h = gelu(h @ Ws[l] + bs[l]) -----------------------
        float* hin  = hb0;
        float* hout = hb1;
        for (int l = 0; l < 4; ++l) {
            const float* hsrc = hin;
            build_A(A_lds, [&](int k) -> float { return hsrc[k]; });
            __syncthreads();
            for (int nt = gwid; nt < 64; nt += TOTWAVES) {
                v8f acc = matvec_tile(Ws_t + (size_t)l * 1048576, nt, lane, A_lds);
                if (lane < 16) {
                    float v = acc[0] + bs[l * H_DIM + nt * 16 + lane];
                    hout[nt * 16 + lane] = gelu_exact(v);
                }
            }
            global_barrier(ctrl);
            float* tmp = hin; hin = hout; hout = tmp;
        }
        // final h is in hin (== hb0)

        // ---- stage E: edge logits + fn logits + rec matvec -----------------
        {
            const float* hsrc = hin;
            build_A(A_lds, [&](int k) -> float { return hsrc[k]; });
            __syncthreads();
        }
        {
            int half = step >> 1;
            int lo   = (half - 1024 > 0) ? (half - 1024) : 0;
            int hi   = 128 + half;
            // jobs: 0..263 edge tiles, 264 fn tile, 265..280 rec tiles
            for (int j = gwid; j < 281; j += TOTWAVES) {
                if (j < 264) {
                    v8f acc = matvec_tile(W_edge_t, j, lane, A_lds);
                    if (lane < 16) {
                        int col = j * 16 + lane;
                        if (col >= lo && col < hi) {
                            float v = acc[0] + b_edge[col] + gumbel_noise(step, 0, col);
                            atomicMax((unsigned long long*)(ctrl + 4), pack_key(v, col));
                        }
                    }
                } else if (j == 264) {
                    if ((step & 1) == 0) {   // fn sampled only on even steps
                        v8f acc = matvec_tile(W_fn_t, 0, lane, A_lds);
                        if (lane < 16) {
                            float v = acc[0] + b_fn[lane] + gumbel_noise(step, 1, lane);
                            atomicMax((unsigned long long*)(ctrl + 6), pack_key(v, lane));
                        }
                    }
                } else {
                    int nt = j - 265;
                    v8f acc = matvec_tile(W_rec_t, nt, lane, A_lds);
                    if (lane < 16) {
                        int col = nt * 16 + lane;
                        float rn = rec_buf[col] + acc[0] + b_rec[col];
                        rec_new[col] = rn;
                        atomicAdd((float*)(ctrl + 8), rn * rn);
                    }
                }
            }
        }
        global_barrier(ctrl);

        // ---- stage F: finalize choices, normalize rec (block 0 only) -------
        if (blockIdx.x == 0) {
            if (threadIdx.x == 0) {
                unsigned long long ek = *(volatile unsigned long long*)(ctrl + 4);
                int e = (int)(0xFFFFFFFFu - (unsigned)(ek & 0xFFFFFFFFull));
                ctrl[2] = (unsigned)e;
                out[step] = (float)e;
                if ((step & 1) == 0) {
                    unsigned long long fk = *(volatile unsigned long long*)(ctrl + 6);
                    int f = (int)(0xFFFFFFFFu - (unsigned)(fk & 0xFFFFFFFFull));
                    ctrl[3] = (unsigned)f;
                    out[L_STEPS + (step >> 1)] = (float)f;
                }
                float ss = *((volatile float*)(ctrl + 8));
                s_scale = 1.0f / fmaxf(sqrtf(ss), 1e-12f);
                *(unsigned long long*)(ctrl + 4) = 0ull;   // reset for next step
                *(unsigned long long*)(ctrl + 6) = 0ull;
                ((float*)(ctrl + 8))[0] = 0.0f;
            }
            __syncthreads();
            float rn = rec_new[threadIdx.x] * s_scale;     // 256 threads
            rec_buf[threadIdx.x] = rn;
            if (step == L_STEPS - 1) out[L_STEPS + L_STEPS / 2 + threadIdx.x] = rn;
        }
        global_barrier(ctrl);
    }
}

// ------------------------- host launch -------------------------------------

extern "C" void kernel_launch(void* const* d_in, const int* in_sizes, int n_in,
                              void* d_out, int out_size, void* d_ws, size_t ws_size,
                              hipStream_t stream) {
    const float* pe       = (const float*)d_in[0];
    const float* W_in     = (const float*)d_in[1];
    const float* b_in     = (const float*)d_in[2];
    const float* Ws       = (const float*)d_in[3];
    const float* bs       = (const float*)d_in[4];
    const float* W_edge   = (const float*)d_in[5];
    const float* b_edge   = (const float*)d_in[6];
    const float* W_fn     = (const float*)d_in[7];
    const float* b_fn     = (const float*)d_in[8];
    const float* W_rec    = (const float*)d_in[9];
    const float* b_rec    = (const float*)d_in[10];
    const float* init_rec = (const float*)d_in[11];

    char*     ws      = (char*)d_ws;
    unsigned* ctrl    = (unsigned*)ws;
    float*    hb0     = (float*)(ws + 256);
    float*    hb1     = hb0 + H_DIM;
    float*    rec_buf = hb1 + H_DIM;
    float*    rec_new = rec_buf + 256;
    __bf16*   Wt      = (__bf16*)(ws + WT_BYTE_OFF);

    const int thr = 256;
    auto cvt = [&](const float* src, __bf16* dst, int N) {
        int total = H_DIM * N;
        convert_tiled<<<(total + thr - 1) / thr, thr, 0, stream>>>(src, dst, N, total);
    };
    cvt(W_in, Wt + OFF_WIN, H_DIM);
    for (int l = 0; l < 4; ++l)
        cvt(Ws + (size_t)l * H_DIM * H_DIM, Wt + OFF_WS + (size_t)l * H_DIM * H_DIM, H_DIM);
    cvt(W_edge, Wt + OFF_WEDGE, E_DIM);
    cvt(W_fn,   Wt + OFF_WFN,   16);
    cvt(W_rec,  Wt + OFF_WREC,  256);

    init_kernel<<<1, 256, 0, stream>>>(ctrl, rec_buf, init_rec);

    nicg_main<<<NWG, NTHR, 0, stream>>>(pe, b_in, bs, b_edge, b_fn, b_rec,
                                        Wt, ctrl, hb0, hb1, rec_buf, rec_new,
                                        (float*)d_out);
}
